// YOLOv3_22608707846970
// MI455X (gfx1250) — matrix operand order repaired and checked by reading, not compiled, not run on hardware
//
#include <hip/hip_runtime.h>
#include <hip/hip_bf16.h>
#include <math.h>

// ---------------------------------------------------------------------------
// Types for CDNA5 WMMA
// ---------------------------------------------------------------------------
typedef _Float16 h16 __attribute__((ext_vector_type(16)));
typedef _Float16 h8  __attribute__((ext_vector_type(8)));
typedef float f8 __attribute__((ext_vector_type(8)));

#define BM 64
#define BN 64
#define BK 32
#define BN_EPS 1e-3f
#define NBOX 10647
#define MAXB 100

// low 32 bits of a generic pointer to __shared__ = LDS byte address
#define LDS_ADDR32(p) ((unsigned)(uintptr_t)(p))

// ---------------------------------------------------------------------------
// Implicit-GEMM conv: out[M=Ho*Wo, Cout] = im2col(in)[M, k*k*Cin] x W[k*k*Cin, Cout]
// f16 in / f16 weights (BN folded) / f32 accumulate via v_wmma_f32_16x16x32_f16.
// 256 threads = 8 wave32; 64x64x32 block tile; each wave owns TWO 16x16 tiles
// -> 2 WMMA per K-step.  Fast path (Cin%32==0): a BK chunk sits inside one
// (ky,kx) tap -> uniform tap index (SALU) and the A tile is staged with
// CDNA5 async copy (global_load_async_to_lds_b128, ASYNCcnt) that writes LDS
// directly, overlapping with the weight-tile staging issued after it.
// ---------------------------------------------------------------------------
__global__ __launch_bounds__(256) void conv_wmma_kernel(
    const _Float16* __restrict__ in,   // [H, W, Cin]
    const _Float16* __restrict__ wgt,  // [k*k*Cin, Cout]   (BN-folded, f16)
    const float*    __restrict__ bias, // [Cout]            (BN-folded, f32)
    _Float16*       __restrict__ out,  // [Ho, Wo, Cout]
    const _Float16* __restrict__ res,  // optional residual add (post-activation)
    int H, int W, int Cin, int Cout, int k, int stride, int leaky)
{
    const int Ho   = (stride == 2) ? (H >> 1) : H;
    const int Wo   = (stride == 2) ? (W >> 1) : W;
    const int M    = Ho * Wo;
    const int Ktot = k * k * Cin;
    const int pad  = (k == 3) ? 1 : 0;
    const bool cinAligned  = (Cin & 31) == 0;
    const bool coutAligned = (Cout & 7) == 0;

    __shared__ _Float16 lsA[BM][BK + 8];  // im2col tile (64 x 32, padded rows)
    __shared__ _Float16 lsB[BN][BK + 8];  // weight tile, TRANSPOSED: [n][kk]

    const int tid  = threadIdx.x;
    const int wave = tid >> 5;
    const int lane = tid & 31;

    const int m0 = blockIdx.x * BM;
    const int n0 = blockIdx.y * BN;

    const int wm = (wave & 3) * 16;    // wave M offset: 0,16,32,48
    const int wn = (wave >> 2) * 32;   // wave N offset: 0,32 (2 subtiles of 16)

    f8 acc[2] = {{}, {}};

    // A staging coords (uniform across K loop)
    const int arow   = tid >> 2;          // 0..63
    const int akloc0 = (tid & 3) * 8;     // 0,8,16,24
    const int am     = m0 + arow;
    const int aoy    = (am < M) ? (am / Wo) : 0;
    const int aox    = (am < M) ? (am % Wo) : 0;
    const unsigned aLds = LDS_ADDR32(&lsA[arow][akloc0]);

    // B staging coords
    const int brow = tid >> 3;            // 0..31 (K row)
    const int bn0  = (tid & 7) * 8;       // 0..56 (N col group of 8)

    #pragma unroll 2
    for (int kb = 0; kb < Ktot; kb += BK) {
        // ---- prefetch next weight tile (CDNA5 global_prefetch path) ----
        if (kb + BK < Ktot)
            __builtin_prefetch((const void*)(wgt + (size_t)(kb + BK + brow) * Cout + n0), 0, 1);

        bool usedAsync = false;

        // ---- stage A (im2col): 64 rows x 32 halves ----
        if (cinAligned) {
            // uniform tap for the whole BK chunk
            int rr = kb / Cin;
            int cb = kb - rr * Cin;
            int ky = rr / k;
            int kx = rr - ky * k;
            int iy = aoy * stride + ky - pad;
            int ix = aox * stride + kx - pad;
            if (am < M && iy >= 0 && iy < H && ix >= 0 && ix < W) {
                // CDNA5 async copy: global -> LDS, 16B per lane, no VGPR bounce
                unsigned long long ga = (unsigned long long)(uintptr_t)
                    (in + ((size_t)iy * W + ix) * Cin + cb + akloc0);
                asm volatile("global_load_async_to_lds_b128 %0, %1, off"
                             :: "v"(aLds), "v"(ga) : "memory");
                usedAsync = true;
            } else {
                *(h8*)(&lsA[arow][akloc0]) = (h8){};  // zero pad rows
            }
        } else {
            #pragma unroll
            for (int e = 0; e < 8; ++e) {
                int kk = kb + akloc0 + e;
                _Float16 v = (_Float16)0.f;
                if (am < M && kk < Ktot) {
                    int c  = kk % Cin;
                    int r  = kk / Cin;
                    int kx = r % k;
                    int ky = r / k;
                    int iy = aoy * stride + ky - pad;
                    int ix = aox * stride + kx - pad;
                    if (iy >= 0 && iy < H && ix >= 0 && ix < W)
                        v = in[((size_t)iy * W + ix) * Cin + c];
                }
                lsA[arow][akloc0 + e] = v;
            }
        }

        // ---- stage B transposed: 32 K-rows x 64 N-cols, 8 halves/thread ----
        // (issued while the async A copy is in flight)
        {
            int kk = kb + brow;
            if (coutAligned && kk < Ktot && (n0 + bn0 + 8) <= Cout) {
                h8 v = *(const h8*)(wgt + (size_t)kk * Cout + n0 + bn0);
                #pragma unroll
                for (int e = 0; e < 8; ++e) lsB[bn0 + e][brow] = v[e];
            } else {
                #pragma unroll
                for (int e = 0; e < 8; ++e) {
                    int n = n0 + bn0 + e;
                    _Float16 v = (_Float16)0.f;
                    if (kk < Ktot && n < Cout) v = wgt[(size_t)kk * Cout + n];
                    lsB[bn0 + e][brow] = v;
                }
            }
        }

        // drain this wave's async LDS writes, then block barrier
        if (usedAsync)
            asm volatile("s_wait_asynccnt 0x0" ::: "memory");
        __syncthreads();

        // ---- A 16x32 / B 32x16 fragments, contiguous k order: k = 16*(lane/16) + s ----
        union { unsigned u[8]; h16 h; } Af, Bf0, Bf1;
        {
            const int mrow  = wm + (lane & 15);
            const int khalf = lane >> 4;  // lanes 16-31 hold K=16..31
            const unsigned* aRow = (const unsigned*)(&lsA[mrow][0]);
            #pragma unroll
            for (int v = 0; v < 8; ++v)
                Af.u[v] = aRow[khalf * 8 + v];

            const int kd = (lane >> 4) * 8;  // dword offset: lanes 16-31 -> K=16..31
            const unsigned* bRow0 = (const unsigned*)(&lsB[wn + (lane & 15)][0]);
            const unsigned* bRow1 = (const unsigned*)(&lsB[wn + 16 + (lane & 15)][0]);
            #pragma unroll
            for (int v = 0; v < 8; ++v) { Bf0.u[v] = bRow0[kd + v]; Bf1.u[v] = bRow1[kd + v]; }
        }

        acc[0] = __builtin_amdgcn_wmma_f32_16x16x32_f16(
            false, Af.h, false, Bf0.h, (short)0, acc[0], false, false);
        acc[1] = __builtin_amdgcn_wmma_f32_16x16x32_f16(
            false, Af.h, false, Bf1.h, (short)0, acc[1], false, false);

        __syncthreads();
    }

    // ---- epilogue: bias + leaky + optional residual, store f16 ----
    #pragma unroll
    for (int t = 0; t < 2; ++t) {
        const int n = n0 + wn + t * 16 + (lane & 15);
        if (n < Cout) {
            const float b = bias[n];
            #pragma unroll
            for (int r = 0; r < 8; ++r) {
                const int m = m0 + wm + r + 8 * (lane >> 4);
                if (m < M) {
                    float y = acc[t][r] + b;
                    if (leaky) y = (y > 0.f) ? y : 0.1f * y;
                    if (res) y += (float)res[(size_t)m * Cout + n];
                    out[(size_t)m * Cout + n] = (_Float16)y;
                }
            }
        }
    }
}

// ---------------------------------------------------------------------------
// BN fold: w' = w * gamma*rsqrt(var+eps) (to f16);  b' = (b-mean)*s + beta
// ---------------------------------------------------------------------------
__global__ void fold_bn_kernel(const float* __restrict__ w, const float* __restrict__ b,
                               const float* __restrict__ gamma, const float* __restrict__ beta,
                               const float* __restrict__ mean, const float* __restrict__ var,
                               _Float16* __restrict__ wf, float* __restrict__ bf,
                               int KC, int Cout)
{
    int i = blockIdx.x * blockDim.x + threadIdx.x;
    if (i < KC) {
        int c = i % Cout;
        float s = gamma[c] * rsqrtf(var[c] + BN_EPS);
        wf[i] = (_Float16)(w[i] * s);
    }
    if (i < Cout) {
        float s = gamma[i] * rsqrtf(var[i] + BN_EPS);
        bf[i] = (b[i] - mean[i]) * s + beta[i];
    }
}

__global__ void fold_plain_kernel(const float* __restrict__ w, const float* __restrict__ b,
                                  _Float16* __restrict__ wf, float* __restrict__ bf,
                                  int KC, int Cout)
{
    int i = blockIdx.x * blockDim.x + threadIdx.x;
    if (i < KC) wf[i] = (_Float16)w[i];
    if (i < Cout) bf[i] = b[i];
}

__global__ void f32_to_f16_kernel(const float* __restrict__ in, _Float16* __restrict__ out, int n)
{
    int i = blockIdx.x * blockDim.x + threadIdx.x;
    if (i < n) out[i] = (_Float16)in[i];
}

// nearest-neighbor 2x upsample into channel slice [coff, coff+C) of a Cout-wide tensor
__global__ void upsample2x_kernel(const _Float16* __restrict__ in, _Float16* __restrict__ out,
                                  int H, int W, int C, int Cout, int coff)
{
    int i = blockIdx.x * blockDim.x + threadIdx.x;
    int N = 4 * H * W * C;
    if (i >= N) return;
    int c  = i % C;
    int t  = i / C;
    int ox = t % (2 * W);
    int oy = t / (2 * W);
    out[((size_t)oy * (2 * W) + ox) * Cout + coff + c] =
        in[((size_t)(oy >> 1) * W + (ox >> 1)) * C + c];
}

// channel-slice copy (for concat)
__global__ void copy_ch_kernel(const _Float16* __restrict__ in, _Float16* __restrict__ out,
                               int HW, int C, int Cout, int coff)
{
    int i = blockIdx.x * blockDim.x + threadIdx.x;
    int N = HW * C;
    if (i >= N) return;
    int c = i % C;
    int p = i / C;
    out[(size_t)p * Cout + coff + c] = in[i];
}

// ---------------------------------------------------------------------------
// YOLO decode: sigmoid/exp box transform + per-box best class score
// ---------------------------------------------------------------------------
__device__ __forceinline__ float sigf(float x) { return 1.f / (1.f + expf(-x)); }

__global__ void decode_kernel(const _Float16* __restrict__ o0, const _Float16* __restrict__ o1,
                              const _Float16* __restrict__ o2,
                              float* __restrict__ boxes, float* __restrict__ scores,
                              int* __restrict__ cls)
{
    int b = blockIdx.x * blockDim.x + threadIdx.x;
    if (b >= NBOX) return;

    const float AN[9][2] = {
        {10.f/416.f,13.f/416.f},{16.f/416.f,30.f/416.f},{33.f/416.f,23.f/416.f},
        {30.f/416.f,61.f/416.f},{62.f/416.f,45.f/416.f},{59.f/416.f,119.f/416.f},
        {116.f/416.f,90.f/416.f},{156.f/416.f,198.f/416.f},{373.f/416.f,326.f/416.f}};

    const _Float16* base; int g, loc, m0;
    if (b < 507)       { base = o0; g = 13; loc = b;        m0 = 6; }
    else if (b < 2535) { base = o1; g = 26; loc = b - 507;  m0 = 3; }
    else               { base = o2; g = 52; loc = b - 2535; m0 = 0; }

    int a    = loc % 3;
    int cell = loc / 3;
    int gx   = cell % g;
    int gy   = cell / g;
    const _Float16* p = base + (size_t)loc * 85;

    float invg = 1.f / (float)g;
    float x = (sigf((float)p[0]) + (float)gx) * invg;
    float y = (sigf((float)p[1]) + (float)gy) * invg;
    float w = expf((float)p[2]) * AN[m0 + a][0];
    float h = expf((float)p[3]) * AN[m0 + a][1];
    float obj = sigf((float)p[4]);

    float best = -1.f; int bi = 0;
    for (int c = 0; c < 80; ++c) {
        float s = obj * sigf((float)p[5 + c]);
        if (s > best) { best = s; bi = c; }
    }
    boxes[(size_t)b * 4 + 0] = x - 0.5f * w;
    boxes[(size_t)b * 4 + 1] = y - 0.5f * h;
    boxes[(size_t)b * 4 + 2] = x + 0.5f * w;
    boxes[(size_t)b * 4 + 3] = y + 0.5f * h;
    scores[b] = best;
    cls[b]    = bi;
}

// ---------------------------------------------------------------------------
// Soft-NMS: 100 serial rounds; single workgroup, wave-parallel argmax + decay.
// sigma=0.5 -> decay = exp(-iou^2). Threshold 0.5.
// ---------------------------------------------------------------------------
__global__ __launch_bounds__(1024) void soft_nms_kernel(
    const float* __restrict__ boxes, float* __restrict__ scores,
    const int* __restrict__ cls,
    float* __restrict__ outB, float* __restrict__ outS,
    int* __restrict__ outC, int* __restrict__ outValid)
{
    __shared__ float sv[1024];
    __shared__ int   si[1024];
    __shared__ float selBox[4];
    __shared__ float selArea;
    __shared__ float selScore;
    __shared__ int   selIdx;
    __shared__ int   selValid;

    const int tid = threadIdx.x;
    int nvalid = 0;

    for (int it = 0; it < MAXB; ++it) {
        // ---- argmax (first-max tie break like jnp.argmax) ----
        float best = -INFINITY; int bidx = 0x7fffffff;
        for (int j = tid; j < NBOX; j += 1024) {
            float v = scores[j];
            if (v > best || (v == best && j < bidx)) { best = v; bidx = j; }
        }
        sv[tid] = best; si[tid] = bidx;
        __syncthreads();
        for (int s = 512; s > 0; s >>= 1) {
            if (tid < s) {
                if (sv[tid + s] > sv[tid] ||
                    (sv[tid + s] == sv[tid] && si[tid + s] < si[tid])) {
                    sv[tid] = sv[tid + s]; si[tid] = si[tid + s];
                }
            }
            __syncthreads();
        }
        if (tid == 0) {
            selIdx   = si[0];
            selScore = sv[0];
            selValid = (sv[0] > 0.5f) ? 1 : 0;
            int ii = selIdx;
            selBox[0] = boxes[(size_t)ii * 4 + 0];
            selBox[1] = boxes[(size_t)ii * 4 + 1];
            selBox[2] = boxes[(size_t)ii * 4 + 2];
            selBox[3] = boxes[(size_t)ii * 4 + 3];
            float aw = selBox[2] - selBox[0]; if (aw < 0.f) aw = 0.f;
            float ah = selBox[3] - selBox[1]; if (ah < 0.f) ah = 0.f;
            selArea = aw * ah;
        }
        __syncthreads();

        if (selValid) {
            for (int j = tid; j < NBOX; j += 1024) {
                float x1 = boxes[(size_t)j * 4 + 0];
                float y1 = boxes[(size_t)j * 4 + 1];
                float x2 = boxes[(size_t)j * 4 + 2];
                float y2 = boxes[(size_t)j * 4 + 3];
                float bw = x2 - x1; if (bw < 0.f) bw = 0.f;
                float bh = y2 - y1; if (bh < 0.f) bh = 0.f;
                float area = bw * bh;
                float ix1 = fmaxf(selBox[0], x1);
                float iy1 = fmaxf(selBox[1], y1);
                float ix2 = fminf(selBox[2], x2);
                float iy2 = fminf(selBox[3], y2);
                float iw = ix2 - ix1; if (iw < 0.f) iw = 0.f;
                float ih = iy2 - iy1; if (ih < 0.f) ih = 0.f;
                float inter = iw * ih;
                float uni   = selArea + area - inter;
                float iou   = (uni > 0.f) ? inter / uni : 0.f;
                scores[j] *= expf(-iou * iou);  // exp(-0.5*iou^2/0.5)
            }
            __syncthreads();
            if (tid == 0) scores[selIdx] = -INFINITY;
        }
        __syncthreads();

        if (tid == 0) {
            int oi = selValid ? selIdx : 0;
            outB[(size_t)it * 4 + 0] = boxes[(size_t)oi * 4 + 0];
            outB[(size_t)it * 4 + 1] = boxes[(size_t)oi * 4 + 1];
            outB[(size_t)it * 4 + 2] = boxes[(size_t)oi * 4 + 2];
            outB[(size_t)it * 4 + 3] = boxes[(size_t)oi * 4 + 3];
            outS[it] = selValid ? selScore : 0.f;
            outC[it] = cls[oi];
            nvalid  += selValid;
        }
        __syncthreads();
    }
    if (tid == 0) outValid[0] = nvalid;
}

// ---------------------------------------------------------------------------
// Host orchestration
// ---------------------------------------------------------------------------
struct LayerDef { int cin, cout, k, bn; };

// insertion (reference-file) order: _BN_LAYERS then _NOBN_LAYERS
enum {
    L_c0 = 0, L_b1_d, L_b1_ra, L_b1_rb, L_b2_d, L_b2_ra, L_b2_rb,
    L_b3_d, L_b3_ra, L_b3_rb, L_b4_d, L_b4_ra, L_b4_rb,
    L_b5_d, L_b5_ra, L_b5_rb,
    L_y0_1, L_y0_2, L_y0_3, L_y0_4, L_y0_5, L_o0_a,
    L_y1_0, L_y1_1, L_y1_2, L_y1_3, L_y1_4, L_y1_5, L_o1_a,
    L_y2_0, L_y2_1, L_y2_2, L_y2_3, L_y2_4, L_y2_5, L_o2_a,
    L_o0_b, L_o1_b, L_o2_b, NLAYERS
};

static const LayerDef LT[NLAYERS] = {
    {3,32,3,1},      {32,64,3,1},   {64,32,1,1},   {32,64,3,1},
    {64,128,3,1},    {128,64,1,1},  {64,128,3,1},
    {128,256,3,1},   {256,128,1,1}, {128,256,3,1},
    {256,512,3,1},   {512,256,1,1}, {256,512,3,1},
    {512,1024,3,1},  {1024,512,1,1},{512,1024,3,1},
    {1024,512,1,1},  {512,1024,3,1},{1024,512,1,1},{512,1024,3,1},{1024,512,1,1},{512,1024,3,1},
    {512,256,1,1},   {768,256,1,1}, {256,512,3,1}, {512,256,1,1}, {256,512,3,1}, {512,256,1,1}, {256,512,3,1},
    {256,128,1,1},   {384,128,1,1}, {128,256,3,1}, {256,128,1,1}, {128,256,3,1}, {256,128,1,1}, {128,256,3,1},
    {1024,255,1,0},  {512,255,1,0}, {256,255,1,0}
};

// alphabetical-name order -> insertion index (for jax pytree sorted-key flattening)
static const int SORTED_PERM[NLAYERS] = {
    1,2,3,4,5,6,7,8,9,10,11,12,13,14,15,       // b1_d..b5_rb
    0,                                          // c0
    21,36,28,37,35,38,                          // o0_a,o0_b,o1_a,o1_b,o2_a,o2_b
    16,17,18,19,20,                             // y0_1..y0_5
    22,23,24,25,26,27,                          // y1_0..y1_5
    29,30,31,32,33,34                           // y2_0..y2_5
};

extern "C" void kernel_launch(void* const* d_in, const int* in_sizes, int n_in,
                              void* d_out, int out_size, void* d_ws, size_t ws_size,
                              hipStream_t stream)
{
    (void)n_in; (void)out_size; (void)ws_size;

    // ---- resolve input pointers (two possible pytree flattening orders) ----
    const float *PW[NLAYERS], *PB[NLAYERS], *PG[NLAYERS], *PBE[NLAYERS], *PM[NLAYERS], *PV[NLAYERS];
    const float* xin = nullptr;
    const int XSZ = 416 * 416 * 3;

    if (in_sizes[0] == XSZ) {
        // insertion order, x first; per-layer key order: w,b,(gamma,beta,mean,var)
        xin = (const float*)d_in[0];
        int c = 1;
        for (int li = 0; li < NLAYERS; ++li) {
            PW[li] = (const float*)d_in[c++];
            PB[li] = (const float*)d_in[c++];
            if (LT[li].bn) {
                PG[li]  = (const float*)d_in[c++];
                PBE[li] = (const float*)d_in[c++];
                PM[li]  = (const float*)d_in[c++];
                PV[li]  = (const float*)d_in[c++];
            } else { PG[li] = PBE[li] = PM[li] = PV[li] = nullptr; }
        }
    } else {
        // jax tree_leaves: dict keys sorted; 'params' < 'x'; per-layer keys sorted:
        // b, beta, gamma, mean, var, w   (no-bn: b, w)
        int c = 0;
        for (int j = 0; j < NLAYERS; ++j) {
            int li = SORTED_PERM[j];
            if (LT[li].bn) {
                PB[li]  = (const float*)d_in[c + 0];
                PBE[li] = (const float*)d_in[c + 1];
                PG[li]  = (const float*)d_in[c + 2];
                PM[li]  = (const float*)d_in[c + 3];
                PV[li]  = (const float*)d_in[c + 4];
                PW[li]  = (const float*)d_in[c + 5];
                c += 6;
            } else {
                PB[li] = (const float*)d_in[c + 0];
                PW[li] = (const float*)d_in[c + 1];
                PG[li] = PBE[li] = PM[li] = PV[li] = nullptr;
                c += 2;
            }
        }
        xin = (const float*)d_in[c];
    }

    // ---- deterministic bump allocator over workspace ----
    size_t off = 0;
    auto alloc = [&](size_t bytes) -> void* {
        void* p = (char*)d_ws + off;
        off += (bytes + 255) & ~(size_t)255;
        return p;
    };

    _Float16* wf16[NLAYERS];
    float*    bf32[NLAYERS];
    for (int li = 0; li < NLAYERS; ++li) {
        size_t KC = (size_t)LT[li].k * LT[li].k * LT[li].cin * LT[li].cout;
        wf16[li] = (_Float16*)alloc(KC * sizeof(_Float16));
        bf32[li] = (float*)alloc((size_t)LT[li].cout * sizeof(float));
    }

    const size_t BUF_E = (size_t)416 * 416 * 32;  // biggest single tensor
    _Float16* bufs[4];
    for (int i = 0; i < 4; ++i) bufs[i] = (_Float16*)alloc(BUF_E * sizeof(_Float16));
    _Float16* bx    = (_Float16*)alloc((size_t)XSZ * sizeof(_Float16));
    _Float16* s36   = (_Float16*)alloc((size_t)52*52*256 * sizeof(_Float16));
    _Float16* s61   = (_Float16*)alloc((size_t)26*26*512 * sizeof(_Float16));
    _Float16* cc1   = (_Float16*)alloc((size_t)26*26*768 * sizeof(_Float16));
    _Float16* cc2   = (_Float16*)alloc((size_t)52*52*384 * sizeof(_Float16));
    _Float16* o0buf = (_Float16*)alloc((size_t)13*13*255 * sizeof(_Float16));
    _Float16* o1buf = (_Float16*)alloc((size_t)26*26*255 * sizeof(_Float16));
    _Float16* o2buf = (_Float16*)alloc((size_t)52*52*255 * sizeof(_Float16));
    float*    boxes = (float*)alloc((size_t)NBOX * 4 * sizeof(float));
    float*    scors = (float*)alloc((size_t)NBOX * sizeof(float));
    int*      clsid = (int*)alloc((size_t)NBOX * sizeof(int));

    // ---- fold BN into f16 weights ----
    for (int li = 0; li < NLAYERS; ++li) {
        int KC = LT[li].k * LT[li].k * LT[li].cin * LT[li].cout;
        int n  = KC > LT[li].cout ? KC : LT[li].cout;
        dim3 g((n + 255) / 256);
        if (LT[li].bn)
            fold_bn_kernel<<<g, 256, 0, stream>>>(PW[li], PB[li], PG[li], PBE[li],
                                                  PM[li], PV[li], wf16[li], bf32[li],
                                                  KC, LT[li].cout);
        else
            fold_plain_kernel<<<g, 256, 0, stream>>>(PW[li], PB[li], wf16[li], bf32[li],
                                                     KC, LT[li].cout);
    }

    // ---- convert input to f16 ----
    f32_to_f16_kernel<<<(XSZ + 255) / 256, 256, 0, stream>>>(xin, bx, XSZ);

    // ---- conv launcher ----
    auto conv = [&](int li, const _Float16* in, _Float16* out, int H, int W,
                    int stride, const _Float16* res) {
        int Ho = (stride == 2) ? (H >> 1) : H;
        int Wo = (stride == 2) ? (W >> 1) : W;
        int M  = Ho * Wo;
        dim3 grid((M + BM - 1) / BM, (LT[li].cout + BN - 1) / BN);
        conv_wmma_kernel<<<grid, 256, 0, stream>>>(
            in, wf16[li], bf32[li], out, res,
            H, W, LT[li].cin, LT[li].cout, LT[li].k, stride, LT[li].bn);
    };

    int cur = 0;
    conv(L_c0, bx, bufs[0], 416, 416, 1, nullptr);  // 416x416x32

    auto down = [&](int li, int H, int W) {
        int d = (cur + 1) & 3;
        conv(li, bufs[cur], bufs[d], H, W, 2, nullptr);
        cur = d;
    };
    auto resblk = [&](int liA, int liB, int n, int H, int W) {
        for (int i = 0; i < n; ++i) {
            int t1 = (cur + 1) & 3, t2 = (cur + 2) & 3;
            conv(liA, bufs[cur], bufs[t1], H, W, 1, nullptr);
            conv(liB, bufs[t1], bufs[t2], H, W, 1, bufs[cur]);
            cur = t2;
        }
    };
    auto step = [&](int li, int H, int W) {
        int t = (cur + 1) & 3;
        conv(li, bufs[cur], bufs[t], H, W, 1, nullptr);
        cur = t;
    };

    down(L_b1_d, 416, 416); resblk(L_b1_ra, L_b1_rb, 1, 208, 208);
    down(L_b2_d, 208, 208); resblk(L_b2_ra, L_b2_rb, 2, 104, 104);
    down(L_b3_d, 104, 104); resblk(L_b3_ra, L_b3_rb, 8, 52, 52);
    hipMemcpyAsync(s36, bufs[cur], (size_t)52*52*256 * sizeof(_Float16),
                   hipMemcpyDeviceToDevice, stream);
    down(L_b4_d, 52, 52);   resblk(L_b4_ra, L_b4_rb, 8, 26, 26);
    hipMemcpyAsync(s61, bufs[cur], (size_t)26*26*512 * sizeof(_Float16),
                   hipMemcpyDeviceToDevice, stream);
    down(L_b5_d, 26, 26);   resblk(L_b5_ra, L_b5_rb, 4, 13, 13);

    // ---- scale 0 head ----
    step(L_y0_1, 13, 13); step(L_y0_2, 13, 13); step(L_y0_3, 13, 13);
    step(L_y0_4, 13, 13); step(L_y0_5, 13, 13);
    {
        int t = (cur + 1) & 3;
        conv(L_o0_a, bufs[cur], bufs[t], 13, 13, 1, nullptr);
        conv(L_o0_b, bufs[t], o0buf, 13, 13, 1, nullptr);
    }

    // ---- scale 1: upsample + concat with x61 ----
    {
        int q = (cur + 2) & 3;
        conv(L_y1_0, bufs[cur], bufs[q], 13, 13, 1, nullptr);  // 13x13x256
        int n1 = 4 * 13 * 13 * 256;
        upsample2x_kernel<<<(n1 + 255) / 256, 256, 0, stream>>>(bufs[q], cc1, 13, 13, 256, 768, 0);
        int n2 = 26 * 26 * 512;
        copy_ch_kernel<<<(n2 + 255) / 256, 256, 0, stream>>>(s61, cc1, 26 * 26, 512, 768, 256);
    }
    conv(L_y1_1, cc1, bufs[0], 26, 26, 1, nullptr); cur = 0;
    step(L_y1_2, 26, 26); step(L_y1_3, 26, 26); step(L_y1_4, 26, 26); step(L_y1_5, 26, 26);
    {
        int t = (cur + 1) & 3;
        conv(L_o1_a, bufs[cur], bufs[t], 26, 26, 1, nullptr);
        conv(L_o1_b, bufs[t], o1buf, 26, 26, 1, nullptr);
    }

    // ---- scale 2: upsample + concat with x36 ----
    {
        int q = (cur + 2) & 3;
        conv(L_y2_0, bufs[cur], bufs[q], 26, 26, 1, nullptr);  // 26x26x128
        int n1 = 4 * 26 * 26 * 128;
        upsample2x_kernel<<<(n1 + 255) / 256, 256, 0, stream>>>(bufs[q], cc2, 26, 26, 128, 384, 0);
        int n2 = 52 * 52 * 256;
        copy_ch_kernel<<<(n2 + 255) / 256, 256, 0, stream>>>(s36, cc2, 52 * 52, 256, 384, 128);
    }
    conv(L_y2_1, cc2, bufs[0], 52, 52, 1, nullptr); cur = 0;
    step(L_y2_2, 52, 52); step(L_y2_3, 52, 52); step(L_y2_4, 52, 52); step(L_y2_5, 52, 52);
    {
        int t = (cur + 1) & 3;
        conv(L_o2_a, bufs[cur], bufs[t], 52, 52, 1, nullptr);
        conv(L_o2_b, bufs[t], o2buf, 52, 52, 1, nullptr);
    }

    // ---- decode + soft-NMS ----
    decode_kernel<<<(NBOX + 255) / 256, 256, 0, stream>>>(o0buf, o1buf, o2buf,
                                                          boxes, scors, clsid);

    float* outB = (float*)d_out;              // [100,4]
    float* outS = outB + MAXB * 4;            // [100]
    int*   outC = (int*)d_out + MAXB * 5;     // [100] int32 bits
    int*   outV = (int*)d_out + MAXB * 6;     // [1]
    soft_nms_kernel<<<1, 1024, 0, stream>>>(boxes, scors, clsid, outB, outS, outC, outV);
}
